// EncoderBlock_44951127720182
// MI455X (gfx1250) — matrix operand order repaired and checked by reading, not compiled, not run on hardware
//
#include <hip/hip_runtime.h>
#include <hip/hip_bf16.h>

// ---------------------------------------------------------------------------
// Transformer encoder block for MI455X (gfx1250, wave32, WMMA + TDM).
// B=2, S=2048, D_MODEL=1024, H=16, d_k=64, D_FF=4096.
// GEMMs: v_wmma_f32_16x16x32_f16, tiles staged LDS via tensor_load_to_lds
// (Tensor Data Mover, double-buffered, TENSORcnt-tracked).
// Attention fused: 32-query-row f32 score strip in LDS (needs CDNA5 320KB/WGP),
// masked softmax in place (coalesced int4 mask reads), attn@V from LDS.
// ---------------------------------------------------------------------------

#define USE_TDM 1

typedef __attribute__((ext_vector_type(16))) _Float16 v16h;
typedef __attribute__((ext_vector_type(8)))  _Float16 v8h;
typedef __attribute__((ext_vector_type(8)))  float    v8f;
typedef __attribute__((ext_vector_type(4)))  int      i32x4;
typedef __attribute__((ext_vector_type(8)))  int      i32x8;

union Frag16 { v16h v; v8h h[2]; };

#define DEV_INLINE __device__ __forceinline__

DEV_INLINE v8f wmma16(v16h a, v16h b, v8f c) {
  // (neg_a, A, neg_b, B, c_mod, C, reuse_a, reuse_b)
  return __builtin_amdgcn_wmma_f32_16x16x32_f16(false, a, false, b, (short)0, c,
                                                false, false);
}

// A fragment 16x32 (MxK). Source row-major, leading dim `ld` (in halves).
DEV_INLINE v16h load_a_frag(const _Float16* base, int ld, int lane) {
  const int l = lane & 15, hp = lane >> 4;
  const _Float16* p = base + (size_t)l * ld + hp * 8;
  Frag16 f;
  f.h[0] = *(const v8h*)(p);
  f.h[1] = *(const v8h*)(p + 16);
  return f.v;
}

// B fragment 32x16 (KxN) sourced from N-major storage (row n, K contiguous).
DEV_INLINE v16h load_b_frag(const _Float16* base, int ld, int lane) {
  const int n = lane & 15, hp = lane >> 4;
  const _Float16* p = base + (size_t)n * ld + hp * 8;
  Frag16 f;
  f.h[0] = *(const v8h*)(p);
  f.h[1] = *(const v8h*)(p + 16);
  return f.v;
}

#if USE_TDM
// ---------------------------------------------------------------------------
// TDM: DMA one 128-row x 32-half tile (row stride K halves) from global into
// LDS at byte offset `lds_off`, padding 16B after every 64B row so the LDS
// row stride is 80B (= LSTR 40 halves, matching the fragment loaders).
// D# group0: count=1, lds_addr, global_addr(57b), type=2.
// D# group1: data_size=2B, pad_enable, pad_interval=16DW, pad_amount=4DW,
//            tensor_dim0=32, tensor_dim1=128, tile 32x128, dim0_stride=K.
// ---------------------------------------------------------------------------
DEV_INLINE void tdm_load_tile(unsigned lds_off, const _Float16* gptr, int ldK) {
  const unsigned long long ga = (unsigned long long)(uintptr_t)gptr;
  i32x4 g0;
  g0[0] = 1;                                   // count=1 (valid user D#)
  g0[1] = (int)lds_off;                        // lds_addr (bytes)
  g0[2] = (int)(unsigned)ga;                   // global_addr[31:0]
  g0[3] = (int)(((unsigned)(ga >> 32) & 0x01ffffffu) | 0x80000000u); // type=2
  i32x8 g1;
  g1[0] = (1 << 16) | (1 << 20) | (3 << 22) | (3 << 25);
  //       data_size=2B  pad_en   intv=16DW    amount=4DW
  g1[1] = (int)(32u << 16);                    // tensor_dim0 = 32 (elements)
  g1[2] = (int)(128u << 16);                   // tensor_dim1 = 128
  g1[3] = (int)(32u << 16);                    // tile_dim0 = 32
  g1[4] = 128;                                 // tile_dim1 = 128
  g1[5] = ldK;                                 // tensor_dim0_stride = K
  g1[6] = 0;
  g1[7] = 0;
  asm volatile("tensor_load_to_lds %0, %1" :: "s"(g0), "s"(g1) : "memory");
}
#endif

// ---------------------------------------------------------------------------
// f32 -> f16 convert, 4 elements / thread, n must be a multiple of 1024.
// ---------------------------------------------------------------------------
__global__ __launch_bounds__(256) void cvt_kernel(const float* __restrict__ in,
                                                  _Float16* __restrict__ out) {
  const int i = (blockIdx.x * 256 + threadIdx.x) * 4;
  const float4 v = *(const float4*)(in + i);
  out[i + 0] = (_Float16)v.x;
  out[i + 1] = (_Float16)v.y;
  out[i + 2] = (_Float16)v.z;
  out[i + 3] = (_Float16)v.w;
}

// ---------------------------------------------------------------------------
// LayerNorm matching torch: unbiased var, eps added to std, scalar gamma/beta.
// ---------------------------------------------------------------------------
__global__ __launch_bounds__(256) void ln_kernel(const float* __restrict__ x,
                                                 const float* __restrict__ gamma,
                                                 const float* __restrict__ beta,
                                                 _Float16* __restrict__ out) {
  constexpr int D = 1024;
  __shared__ float red[256];
  const int tid = threadIdx.x;
  const size_t row = blockIdx.x;
  const float* xr = x + row * D;
  const float4 v = ((const float4*)xr)[tid];

  red[tid] = v.x + v.y + v.z + v.w;
  __syncthreads();
  for (int off = 128; off > 0; off >>= 1) {
    if (tid < off) red[tid] += red[tid + off];
    __syncthreads();
  }
  const float mean = red[0] * (1.0f / D);
  __syncthreads();

  const float dx = v.x - mean, dy = v.y - mean, dz = v.z - mean, dw = v.w - mean;
  red[tid] = dx * dx + dy * dy + dz * dz + dw * dw;
  __syncthreads();
  for (int off = 128; off > 0; off >>= 1) {
    if (tid < off) red[tid] += red[tid + off];
    __syncthreads();
  }
  const float var = red[0] * (1.0f / (D - 1));
  const float inv = gamma[0] / (sqrtf(var) + 1e-5f);
  const float bb = beta[0];

  _Float16* orow = out + row * D;
  orow[tid * 4 + 0] = (_Float16)(dx * inv + bb);
  orow[tid * 4 + 1] = (_Float16)(dy * inv + bb);
  orow[tid * 4 + 2] = (_Float16)(dz * inv + bb);
  orow[tid * 4 + 3] = (_Float16)(dw * inv + bb);
}

// ---------------------------------------------------------------------------
// Tiled WMMA GEMM: C[M,N] = act(A[M,K] @ W[N,K]^T + bias[N]) (+ resid).
// Block tile 128x128, 8 waves (4xM, 2xN), wave tile 32x64, K-step 32.
// Tiles staged via TDM (double-buffered, one issuing wave, TENSORcnt pipelined)
// EPI: 0 = bias -> f16   1 = bias -> f16 transposed (ldT)
//      2 = bias+ReLU -> f16   3 = bias+residual -> f32
// M,N multiples of 128; K multiple of 32.
// ---------------------------------------------------------------------------
template <int EPI>
__global__ __launch_bounds__(256) void gemm_kernel(
    const _Float16* __restrict__ A, const _Float16* __restrict__ Bw,
    const float* __restrict__ bias, const float* __restrict__ resid,
    void* __restrict__ out, int M, int N, int K, int ldT) {
  constexpr int LSTR = 40;  // halves; 80B LDS row stride (64B data + 16B pad)
  __shared__ _Float16 As[2][128 * LSTR];
  __shared__ _Float16 Bs[2][128 * LSTR];

  const int tid = threadIdx.x;
  const int lane = tid & 31;
  const int wid = tid >> 5;
  const int wm = wid & 3;   // wave row (0..3) -> 32 rows each
  const int wn = wid >> 2;  // wave col (0..1) -> 64 cols each
  const int m0 = blockIdx.y * 128;
  const int n0 = blockIdx.x * 128;
  const int nsteps = K >> 5;

  const v8f zero = {0.f, 0.f, 0.f, 0.f, 0.f, 0.f, 0.f, 0.f};
  v8f acc[2][4];
#pragma unroll
  for (int i = 0; i < 2; ++i)
#pragma unroll
    for (int j = 0; j < 4; ++j) acc[i][j] = zero;

#if USE_TDM
  const int issuer = (__builtin_amdgcn_readfirstlane((int)tid) >> 5) == 0;
  const unsigned ldsA[2] = {(unsigned)(uintptr_t)&As[0][0],
                            (unsigned)(uintptr_t)&As[1][0]};
  const unsigned ldsB[2] = {(unsigned)(uintptr_t)&Bs[0][0],
                            (unsigned)(uintptr_t)&Bs[1][0]};
  if (issuer) {  // prologue: DMA k-tile 0 into buffer 0
    tdm_load_tile(ldsA[0], A + (size_t)m0 * K, K);
    tdm_load_tile(ldsB[0], Bw + (size_t)n0 * K, K);
  }
#endif

  for (int i = 0; i < nsteps; ++i) {
    const int cur = i & 1;
#if USE_TDM
    if (issuer) {
      if (i + 1 < nsteps) {  // overlap: DMA next tile, wait only for current
        const int nk = (i + 1) * 32;
        tdm_load_tile(ldsA[cur ^ 1], A + (size_t)m0 * K + nk, K);
        tdm_load_tile(ldsB[cur ^ 1], Bw + (size_t)n0 * K + nk, K);
        __builtin_amdgcn_s_wait_tensorcnt(2);
      } else {
        __builtin_amdgcn_s_wait_tensorcnt(0);
      }
    }
#else
    {  // fallback: per-thread staging (2 threads per row)
      const int k0 = i * 32;
      const int r = tid >> 1;
      const int c = (tid & 1) * 16;
      const _Float16* ga = A + (size_t)(m0 + r) * K + k0 + c;
      *(v8h*)&As[cur][r * LSTR + c]     = *(const v8h*)(ga);
      *(v8h*)&As[cur][r * LSTR + c + 8] = *(const v8h*)(ga + 8);
      const _Float16* gb = Bw + (size_t)(n0 + r) * K + k0 + c;
      *(v8h*)&Bs[cur][r * LSTR + c]     = *(const v8h*)(gb);
      *(v8h*)&Bs[cur][r * LSTR + c + 8] = *(const v8h*)(gb + 8);
    }
#endif
    __syncthreads();

    const _Float16* Ab = &As[cur][0];
    const _Float16* Bb = &Bs[cur][0];
    v16h af[2], bf[4];
#pragma unroll
    for (int mi = 0; mi < 2; ++mi)
      af[mi] = load_a_frag(Ab + (wm * 32 + mi * 16) * LSTR, LSTR, lane);
#pragma unroll
    for (int ni = 0; ni < 4; ++ni)
      bf[ni] = load_b_frag(Bb + (wn * 64 + ni * 16) * LSTR, LSTR, lane);
#pragma unroll
    for (int mi = 0; mi < 2; ++mi)
#pragma unroll
      for (int ni = 0; ni < 4; ++ni)
        acc[mi][ni] = wmma16(af[mi], bf[ni], acc[mi][ni]);
    __syncthreads();
  }

  // Epilogue. C/D layout: row = e + 8*hp, col = lane&15 per 16x16 frag.
  const int l16 = lane & 15, hp = lane >> 4;
#pragma unroll
  for (int mi = 0; mi < 2; ++mi) {
#pragma unroll
    for (int ni = 0; ni < 4; ++ni) {
      const int col = n0 + wn * 64 + ni * 16 + l16;
      const float bv = bias[col];
#pragma unroll
      for (int e = 0; e < 8; ++e) {
        const int row = m0 + wm * 32 + mi * 16 + hp * 8 + e;
        float val = acc[mi][ni][e] + bv;
        if constexpr (EPI == 2) val = fmaxf(val, 0.0f);
        if constexpr (EPI == 0 || EPI == 2) {
          ((_Float16*)out)[(size_t)row * N + col] = (_Float16)val;
        } else if constexpr (EPI == 1) {
          ((_Float16*)out)[(size_t)col * ldT + row] = (_Float16)val;
        } else {
          ((float*)out)[(size_t)row * N + col] =
              val + resid[(size_t)row * N + col];
        }
      }
    }
  }
}

// ---------------------------------------------------------------------------
// Fused attention. One block = one (b, h, 32-query-row strip).
// Phase 1: scores = Q K^T / 8 (WMMA) -> 32 x (2048+8) f32 LDS strip.
// Phase 2: mask (coalesced int4 reads) + row softmax in LDS, float4 I/O.
// Phase 3: ctx = attn @ V (WMMA), A-frags converted f32->f16 from LDS.
// ---------------------------------------------------------------------------
__global__ __launch_bounds__(256) void attn_kernel(
    const _Float16* __restrict__ q16, const _Float16* __restrict__ k16,
    const _Float16* __restrict__ vt16, const int* __restrict__ mask,
    _Float16* __restrict__ ctx16) {
  constexpr int S = 2048, DM = 1024, dk = 64, BS = 4096;  // BS = B*S
  constexpr int SROW = S + 8;                             // padded f32 stride
  extern __shared__ float sc[];                           // 32 x SROW
  __shared__ float rs[256];

  const int tid = threadIdx.x, lane = tid & 31, w = tid >> 5;
  const int l = lane & 15, hp = lane >> 4;
  const int blk = blockIdx.x;
  const int qb = blk & 63;
  const int h = (blk >> 6) & 15;
  const int b = blk >> 10;
  const size_t bS = (size_t)b * S;
  const int q0 = qb * 32;
  const v8f zero = {0.f, 0.f, 0.f, 0.f, 0.f, 0.f, 0.f, 0.f};

  // ---- phase 1: scaled scores into LDS (mask deferred to phase 2) ----
  v16h af[2][2];
#pragma unroll
  for (int mi = 0; mi < 2; ++mi)
#pragma unroll
    for (int ks = 0; ks < 2; ++ks)
      af[mi][ks] =
          load_a_frag(q16 + (bS + q0 + mi * 16) * DM + h * dk + ks * 32, DM, lane);

  const int nw0 = w * 256;  // each wave owns 256 key columns
#pragma unroll 1
  for (int ni = 0; ni < 16; ++ni) {
    const int n0 = nw0 + ni * 16;
    v8f a0 = zero, a1 = zero;
#pragma unroll
    for (int ks = 0; ks < 2; ++ks) {
      const v16h bf =
          load_b_frag(k16 + (bS + n0) * DM + h * dk + ks * 32, DM, lane);
      a0 = wmma16(af[0][ks], bf, a0);
      a1 = wmma16(af[1][ks], bf, a1);
    }
    const int col = n0 + l;
#pragma unroll
    for (int e = 0; e < 8; ++e) {
      sc[(hp * 8 + e) * SROW + col] = a0[e] * 0.125f;
      sc[(16 + hp * 8 + e) * SROW + col] = a1[e] * 0.125f;
    }
  }
  __syncthreads();

  // ---- phase 2: mask + softmax, 8 threads x 256 contiguous cols per row ----
  {
    const int row = tid >> 3, sg = tid & 7;
    const int4* mp = (const int4*)(mask + (size_t)(q0 + row) * S + sg * 256);
    float* rp = sc + row * SROW + sg * 256;

    float mx = -3.0e38f;
    for (int j = 0; j < 64; ++j) {
      const int4 m4 = mp[j];
      float4 s4 = *(const float4*)(rp + 4 * j);
      s4.x = (m4.x == 0) ? -1e9f : s4.x;
      s4.y = (m4.y == 0) ? -1e9f : s4.y;
      s4.z = (m4.z == 0) ? -1e9f : s4.z;
      s4.w = (m4.w == 0) ? -1e9f : s4.w;
      *(float4*)(rp + 4 * j) = s4;
      mx = fmaxf(mx, fmaxf(fmaxf(s4.x, s4.y), fmaxf(s4.z, s4.w)));
    }
    rs[tid] = mx; __syncthreads();
    if (sg < 4) rs[tid] = fmaxf(rs[tid], rs[tid + 4]); __syncthreads();
    if (sg < 2) rs[tid] = fmaxf(rs[tid], rs[tid + 2]); __syncthreads();
    if (sg == 0) rs[tid] = fmaxf(rs[tid], rs[tid + 1]); __syncthreads();
    mx = rs[row * 8]; __syncthreads();

    float sm = 0.f;
    for (int j = 0; j < 64; ++j) {
      float4 s4 = *(const float4*)(rp + 4 * j);
      s4.x = __expf(s4.x - mx);
      s4.y = __expf(s4.y - mx);
      s4.z = __expf(s4.z - mx);
      s4.w = __expf(s4.w - mx);
      *(float4*)(rp + 4 * j) = s4;
      sm += s4.x + s4.y + s4.z + s4.w;
    }
    rs[tid] = sm; __syncthreads();
    if (sg < 4) rs[tid] += rs[tid + 4]; __syncthreads();
    if (sg < 2) rs[tid] += rs[tid + 2]; __syncthreads();
    if (sg == 0) rs[tid] += rs[tid + 1]; __syncthreads();
    const float inv = 1.0f / rs[row * 8];
    for (int j = 0; j < 64; ++j) {
      float4 s4 = *(const float4*)(rp + 4 * j);
      s4.x *= inv; s4.y *= inv; s4.z *= inv; s4.w *= inv;
      *(float4*)(rp + 4 * j) = s4;
    }
  }
  __syncthreads();

  // ---- phase 3: ctx = attn @ V; one 16x16 output frag per wave (2x4) ----
  {
    const int mi = w & 1, n4 = w >> 1;
    v8f acc = zero;
    for (int ks = 0; ks < S / 32; ++ks) {
      Frag16 fa;
      const float* ap = sc + (mi * 16 + l) * SROW + ks * 32;
#pragma unroll
      for (int j = 0; j < 8; ++j) fa.v[j] = (_Float16)ap[hp * 8 + j];
#pragma unroll
      for (int j = 0; j < 8; ++j) fa.v[8 + j] = (_Float16)ap[16 + hp * 8 + j];
      const v16h fb = load_b_frag(
          vt16 + (size_t)(h * dk + n4 * 16) * BS + bS + ks * 32, BS, lane);
      acc = wmma16(fa.v, fb, acc);
    }
#pragma unroll
    for (int e = 0; e < 8; ++e) {
      const size_t row = bS + q0 + mi * 16 + hp * 8 + e;
      ctx16[row * DM + h * dk + n4 * 16 + l] = (_Float16)acc[e];
    }
  }
}

// ---------------------------------------------------------------------------
// Host-side orchestration (all on `stream`, graph-capture safe).
// Workspace (~88 MiB):
//   [ 0, 2)MiB wq16   [ 2, 4) wk16   [ 4, 6) wv16   [ 6, 8) wo16
//   [ 8,16) w1_16     [16,24) w2_16
//   [24,32) xn16  (reused as ctx16)      [32,40) q16 (reused as xn2)
//   [40,48) k16   [48,56) vt16  -> [40,72) reused as h16 (FFN hidden)
//   [72,88) xres (f32 residual-1 output)
// ---------------------------------------------------------------------------
extern "C" void kernel_launch(void* const* d_in, const int* in_sizes, int n_in,
                              void* d_out, int out_size, void* d_ws,
                              size_t ws_size, hipStream_t stream) {
  (void)in_sizes; (void)n_in; (void)out_size; (void)ws_size;

  const float* x    = (const float*)d_in[0];
  const int*   mask = (const int*)d_in[1];
  const float* wq = (const float*)d_in[2];
  const float* bq = (const float*)d_in[3];
  const float* wk = (const float*)d_in[4];
  const float* bk = (const float*)d_in[5];
  const float* wv = (const float*)d_in[6];
  const float* bv = (const float*)d_in[7];
  const float* wo = (const float*)d_in[8];
  const float* bo = (const float*)d_in[9];
  const float* w1 = (const float*)d_in[10];
  const float* b1 = (const float*)d_in[11];
  const float* w2 = (const float*)d_in[12];
  const float* b2 = (const float*)d_in[13];
  const float* g1  = (const float*)d_in[14];
  const float* be1 = (const float*)d_in[15];
  const float* g2  = (const float*)d_in[16];
  const float* be2 = (const float*)d_in[17];
  float* out = (float*)d_out;

  char* ws = (char*)d_ws;
  constexpr size_t MiB = 1024 * 1024;
  _Float16* wq16 = (_Float16*)(ws + 0 * MiB);
  _Float16* wk16 = (_Float16*)(ws + 2 * MiB);
  _Float16* wv16 = (_Float16*)(ws + 4 * MiB);
  _Float16* wo16 = (_Float16*)(ws + 6 * MiB);
  _Float16* w116 = (_Float16*)(ws + 8 * MiB);
  _Float16* w216 = (_Float16*)(ws + 16 * MiB);
  _Float16* xn16 = (_Float16*)(ws + 24 * MiB);
  _Float16* q16  = (_Float16*)(ws + 32 * MiB);
  _Float16* k16  = (_Float16*)(ws + 40 * MiB);
  _Float16* vt16 = (_Float16*)(ws + 48 * MiB);
  float*    xres = (float*)(ws + 72 * MiB);
  _Float16* ctx16 = xn16;  // alias: xn16 dead after V projection
  _Float16* xn2   = q16;   // alias: q16 dead after attention
  _Float16* h16   = k16;   // alias: [40,72)MiB dead after attention

  const dim3 blk(256);

  // one-time weight f16 conversion (n/1024 blocks, 4 elems/thread)
  cvt_kernel<<<1024, blk, 0, stream>>>(wq, wq16);
  cvt_kernel<<<1024, blk, 0, stream>>>(wk, wk16);
  cvt_kernel<<<1024, blk, 0, stream>>>(wv, wv16);
  cvt_kernel<<<1024, blk, 0, stream>>>(wo, wo16);
  cvt_kernel<<<4096, blk, 0, stream>>>(w1, w116);
  cvt_kernel<<<4096, blk, 0, stream>>>(w2, w216);

  // residual 1: pre-LN attention
  ln_kernel<<<4096, blk, 0, stream>>>(x, g1, be1, xn16);
  gemm_kernel<0><<<dim3(8, 32), blk, 0, stream>>>(xn16, wq16, bq, nullptr, q16,
                                                  4096, 1024, 1024, 0);
  gemm_kernel<0><<<dim3(8, 32), blk, 0, stream>>>(xn16, wk16, bk, nullptr, k16,
                                                  4096, 1024, 1024, 0);
  gemm_kernel<1><<<dim3(8, 32), blk, 0, stream>>>(xn16, wv16, bv, nullptr, vt16,
                                                  4096, 1024, 1024, 4096);
  attn_kernel<<<2048, blk, (size_t)32 * (2048 + 8) * sizeof(float), stream>>>(
      q16, k16, vt16, mask, ctx16);
  gemm_kernel<3><<<dim3(8, 32), blk, 0, stream>>>(ctx16, wo16, bo, x, xres,
                                                  4096, 1024, 1024, 0);

  // residual 2: pre-LN feed-forward
  ln_kernel<<<4096, blk, 0, stream>>>(xres, g2, be2, xn2);
  gemm_kernel<2><<<dim3(32, 32), blk, 0, stream>>>(xn2, w116, b1, nullptr, h16,
                                                   4096, 4096, 1024, 0);
  gemm_kernel<3><<<dim3(8, 32), blk, 0, stream>>>(h16, w216, b2, xres, out,
                                                  4096, 1024, 4096, 0);
}
